// LSTM_3985729650728
// MI455X (gfx1250) — matrix operand order repaired and checked by reading, not compiled, not run on hardware
//
#include <hip/hip_runtime.h>

#define TT 512
#define BB 64
#define HH 512
#define LL 2
#define H4 2048

typedef __attribute__((ext_vector_type(16))) __bf16 v16bf;
typedef __attribute__((ext_vector_type(8)))  float  v8f;
typedef __attribute__((ext_vector_type(4)))  int    v4i;

// explicit global-address-space pointer: keeps laundered loads on the
// GLOBAL path (global_load_b128, LOADcnt only) instead of FLAT
// (which would consume both LOADcnt and DScnt and the LDS issue path).
typedef const __attribute__((address_space(1))) v4i* gcv4i;

union Frag {
  v4i   q[2];
  v16bf f;
};

__device__ __forceinline__ unsigned f2bf(float x) {
  unsigned u = __float_as_uint(x);
  return (u + 0x7FFFu + ((u >> 16) & 1u)) >> 16;   // round-to-nearest-even
}

__device__ __forceinline__ float sigf(float x) { return 1.0f / (1.0f + __expf(-x)); }

// ---------------------------------------------------------------------------
// Pack fp32 weights (Wx||Wh concatenated along K -> 1024) into bf16 in the
// exact WMMA B-fragment register order:
//   dword index = layer*2^20 + ((tile*32 + chunk)*32 + lane)*8 + j
// B 32x16, contiguous k order used here (A matches): VGPR j holds K = 2j,2j+1 (lanes 0-15) or K = 16+2j,17+2j
// (lanes 16-31), column N = 16*tile + (lane&15).
// ---------------------------------------------------------------------------
__global__ void pack_weights_kernel(const float* __restrict__ Wh,
                                    const float* __restrict__ Wx,
                                    unsigned* __restrict__ pk) {
  unsigned idx = blockIdx.x * blockDim.x + threadIdx.x;
  if (idx >= (2u << 20)) return;
  unsigned j     = idx & 7u;
  unsigned lane  = (idx >> 3) & 31u;
  unsigned c     = (idx >> 8) & 31u;
  unsigned t     = (idx >> 13) & 127u;
  unsigned layer = idx >> 20;
  unsigned n     = t * 16u + (lane & 15u);
  unsigned koff  = ((lane >> 4) << 4) + 2u * j;
  const float* W;
  unsigned k;
  if (c < 16u) { W = Wx + (size_t)layer * HH * H4; k = 32u * c + koff; }
  else         { W = Wh + (size_t)layer * HH * H4; k = 32u * (c - 16u) + koff; }
  unsigned lo = f2bf(W[(size_t)k * H4 + n]);
  unsigned hi = f2bf(W[(size_t)(k + 1u) * H4 + n]);
  pk[idx] = lo | (hi << 16);
}

__global__ void pack_x_kernel(const float* __restrict__ x,
                              unsigned* __restrict__ xb) {
  unsigned idx = blockIdx.x * blockDim.x + threadIdx.x;
  if (idx >= (unsigned)(TT * BB * HH / 2)) return;
  unsigned lo = f2bf(x[2 * (size_t)idx]);
  unsigned hi = f2bf(x[2 * (size_t)idx + 1]);
  xb[idx] = lo | (hi << 16);
}

// ---------------------------------------------------------------------------
// One GEMM phase: 16 K-chunks of acc[g] += A*B[g] for this wave's 4 gate
// tiles. All B addresses = per-lane base + compile-time immediate:
//   B byte offset = LAYER*4MB + g*1MB + (PHASE+c)*1KB (+16)
// The base is laundered through an empty volatile asm so LICM can never
// hoist the (t-invariant) weight loads out of the sequence loop, then cast
// to an addrspace(1) pointer so they stay global_load_b128 (LOADcnt only).
// B is double-buffered in registers; A double-buffered only when from
// global x (AX), single-buffered when from LDS h.
// ---------------------------------------------------------------------------
template <int LAYER, int PHASE, bool AX>
__device__ __forceinline__ void gemm_phase(v8f acc[4],
                                           const char* pkb,
                                           const char* __restrict__ ax,
                                           const unsigned short* __restrict__ ah) {
  // opacify the weight base: loads below become loop-variant, un-hoistable
  unsigned long long pbv = (unsigned long long)pkb;
  asm volatile("" : "+v"(pbv));

  Frag a[2];
  Frag b[2][4];

  auto ldb = [&](Frag* dst, int c) {
#pragma unroll
    for (int g = 0; g < 4; ++g) {
      unsigned long long p = pbv + (size_t)LAYER * (4u << 20)
                                 + (size_t)g * (1u << 20)
                                 + (size_t)(PHASE + c) * 1024;
      dst[g].q[0] = *(gcv4i)(p);
      dst[g].q[1] = *(gcv4i)(p + 16);
    }
  };
  auto lda = [&](Frag& d, int c) {
    if constexpr (AX) {
      const char* p = ax + c * 64;           // global bf16 x row
      d.q[0] = *(const v4i*)(p);
      d.q[1] = *(const v4i*)(p + 16);
    } else {
      const unsigned short* p = ah + c * 32; // LDS h row
      d.q[0] = *(const v4i*)(p);
      d.q[1] = *(const v4i*)(p + 8);
    }
  };

  ldb(b[0], 0);
  if constexpr (AX) lda(a[0], 0);

#pragma unroll
  for (int c = 0; c < 16; ++c) {
    const int cur = c & 1;
    const int nxt = cur ^ 1;
    if constexpr (AX) {
      if (c < 15) lda(a[nxt], c + 1);
    } else {
      lda(a[0], c);
    }
    if (c < 15) ldb(b[nxt], c + 1);
    const Frag& af = AX ? a[cur] : a[0];
#pragma unroll
    for (int g = 0; g < 4; ++g)
      acc[g] = __builtin_amdgcn_wmma_f32_16x16x32_bf16(
          false, af.f, false, b[cur][g].f, (short)0, acc[g], false, false);
  }
}

// ---------------------------------------------------------------------------
// Persistent LSTM. One workgroup (1024 thr = 32 waves) per 16 batch rows;
// batch-independent recurrence -> no inter-workgroup sync. Wave w owns gate
// tiles {w, w+32, w+64, w+96} = i/f/g/o for hidden cols [16w,16w+16).
// h (bf16) exchanged via LDS; cell state + bias live in LDS to keep VGPRs
// free for the double-buffered WMMA pipeline.
// ---------------------------------------------------------------------------
__launch_bounds__(1024, 1)
__global__ void lstm_kernel(const unsigned* __restrict__ pk,
                            const unsigned* __restrict__ xb,
                            const float* __restrict__ bh,
                            float* __restrict__ out) {
  __shared__ unsigned short lds_h[LL][16 * HH];   // 32 KB
  __shared__ float lds_c[LL][8][1024];            // 64 KB, lane-contiguous
  __shared__ float lds_bias[LL * H4];             // 16 KB

  const int tid   = threadIdx.x;
  const int lane  = tid & 31;
  const int wave  = tid >> 5;
  const int nlo   = lane & 15;
  const int hi16  = lane >> 4;
  const int brow0 = blockIdx.x * 16;
  const int col   = wave * 16 + nlo;

  for (int i = tid; i < LL * 16 * HH; i += 1024) ((unsigned short*)lds_h)[i] = 0;
  for (int i = tid; i < LL * 8 * 1024; i += 1024) ((float*)lds_c)[i] = 0.f;
  for (int i = tid; i < LL * H4; i += 1024)       lds_bias[i] = bh[i];
  __syncthreads();

  // whole-kernel per-lane B base: all B loads become base + immediate
  const char* pkb = (const char*)pk + wave * 32768 + lane * 32;
  // LDS A row bases (element ptrs, hi-half fold: +32 bytes = +16 shorts)
  const unsigned short* ah0 = &lds_h[0][nlo * HH] + hi16 * 16;
  const unsigned short* ah1 = &lds_h[1][nlo * HH] + hi16 * 16;

  float* hfin = out + (size_t)TT * BB * HH;
  float* cfin = hfin + (size_t)LL * BB * HH;

  for (int t = 0; t < TT; ++t) {
    const char* xr = (const char*)xb
        + (size_t)(t * BB + brow0 + nlo) * 1024 + hi16 * 32;

    // ======================= layer 0 =======================
    {
      v8f acc[4];
#pragma unroll
      for (int g = 0; g < 4; ++g) acc[g] = (v8f){0, 0, 0, 0, 0, 0, 0, 0};
      gemm_phase<0, 0,  true >(acc, pkb, xr, nullptr);   // x  @ Wx
      gemm_phase<0, 16, false>(acc, pkb, nullptr, ah0);  // h0 @ Wh

      float b0 = lds_bias[0 * H4 + 0 * HH + col];
      float b1 = lds_bias[0 * H4 + 1 * HH + col];
      float b2 = lds_bias[0 * H4 + 2 * HH + col];
      float b3 = lds_bias[0 * H4 + 3 * HH + col];
      float hv[8];
#pragma unroll
      for (int i = 0; i < 8; ++i) {
        float ig = sigf(acc[0][i] + b0);
        float fg = sigf(acc[1][i] + b1);
        float gg = tanhf(acc[2][i] + b2);
        float og = sigf(acc[3][i] + b3);
        float cv = fg * lds_c[0][i][tid] + ig * gg;
        lds_c[0][i][tid] = cv;
        hv[i] = og * tanhf(cv);
      }
      __syncthreads();                  // all waves done reading lds_h
#pragma unroll
      for (int i = 0; i < 8; ++i) {
        int row = i + hi16 * 8;
        lds_h[0][row * HH + col] = (unsigned short)f2bf(hv[i]);
        if (t == TT - 1)
          hfin[((size_t)0 * BB + brow0 + row) * HH + col] = hv[i];
      }
      __syncthreads();
    }

    // ======================= layer 1 =======================
    {
      v8f acc[4];
#pragma unroll
      for (int g = 0; g < 4; ++g) acc[g] = (v8f){0, 0, 0, 0, 0, 0, 0, 0};
      gemm_phase<1, 0,  false>(acc, pkb, nullptr, ah0);  // h0 @ Wx
      gemm_phase<1, 16, false>(acc, pkb, nullptr, ah1);  // h1 @ Wh

      float b0 = lds_bias[1 * H4 + 0 * HH + col];
      float b1 = lds_bias[1 * H4 + 1 * HH + col];
      float b2 = lds_bias[1 * H4 + 2 * HH + col];
      float b3 = lds_bias[1 * H4 + 3 * HH + col];
      float hv[8];
#pragma unroll
      for (int i = 0; i < 8; ++i) {
        float ig = sigf(acc[0][i] + b0);
        float fg = sigf(acc[1][i] + b1);
        float gg = tanhf(acc[2][i] + b2);
        float og = sigf(acc[3][i] + b3);
        float cv = fg * lds_c[1][i][tid] + ig * gg;
        lds_c[1][i][tid] = cv;
        hv[i] = og * tanhf(cv);
      }
      __syncthreads();
#pragma unroll
      for (int i = 0; i < 8; ++i) {
        int row = i + hi16 * 8;
        lds_h[1][row * HH + col] = (unsigned short)f2bf(hv[i]);
        out[((size_t)t * BB + brow0 + row) * HH + col] = hv[i];
        if (t == TT - 1)
          hfin[((size_t)1 * BB + brow0 + row) * HH + col] = hv[i];
      }
      __syncthreads();
    }
  }

  // c_fin from LDS-resident cell state (exact fp32)
#pragma unroll
  for (int layer = 0; layer < LL; ++layer)
#pragma unroll
    for (int i = 0; i < 8; ++i) {
      int row = i + hi16 * 8;
      cfin[((size_t)layer * BB + brow0 + row) * HH + col] = lds_c[layer][i][tid];
    }
}

extern "C" void kernel_launch(void* const* d_in, const int* in_sizes, int n_in,
                              void* d_out, int out_size, void* d_ws, size_t ws_size,
                              hipStream_t stream) {
  const float* x  = (const float*)d_in[0];
  const float* Wh = (const float*)d_in[1];
  const float* Wx = (const float*)d_in[2];
  const float* bh = (const float*)d_in[3];

  // workspace: [0,8MB) packed bf16 weights, [8MB,40MB) packed bf16 x
  unsigned* pk = (unsigned*)d_ws;
  unsigned* xb = pk + (2u << 20);

  {
    unsigned n = 2u << 20;
    pack_weights_kernel<<<dim3(n / 256), dim3(256), 0, stream>>>(Wh, Wx, pk);
  }
  {
    unsigned n = (unsigned)(TT * BB * HH / 2);
    pack_x_kernel<<<dim3((n + 255) / 256), dim3(256), 0, stream>>>(x, xb);
  }
  lstm_kernel<<<dim3(BB / 16), dim3(1024), 0, stream>>>(pk, xb, bh, (float*)d_out);
}
